// ModuleM_49752901157428
// MI455X (gfx1250) — matrix operand order repaired and checked by reading, not compiled, not run on hardware
//
#include <hip/hip_runtime.h>
#include <hip/hip_bf16.h>

// ---------------------------------------------------------------------------
// Stereo refinement net on MI455X (gfx1250): warp + cost volume + 5 WMMA conv
// layers (f16 data, f32 accum, double-buffered LDS pipeline) + scalar last conv.
// ---------------------------------------------------------------------------

typedef __attribute__((ext_vector_type(16))) _Float16 v16h;
typedef __attribute__((ext_vector_type(8)))  _Float16 v8h;
typedef __attribute__((ext_vector_type(8)))  float    v8f;
typedef __attribute__((ext_vector_type(4)))  float    v4f;

#define IMG_B 4
#define IMG_H 192
#define IMG_W 640
#define NPIX  (IMG_B * IMG_H * IMG_W)   // 491520

// ---------------------------------------------------------------------------
// Kernel 1: bilinear-upsample prev_disp, warp right image, write:
//   act0[..., 0:64]  = left (f16)
//   act0[..., 64:96] = 0 (cost channels filled later, rest stays zero padding)
//   warped[..., 0:64] = warped right (f16)
// ---------------------------------------------------------------------------
__global__ __launch_bounds__(256)
void prep_warp(const float* __restrict__ left,
               const float* __restrict__ right,
               const float* __restrict__ prev,
               _Float16* __restrict__ act0,     // stride 96
               _Float16* __restrict__ warped)   // stride 64
{
    int idx = blockIdx.x * 256 + threadIdx.x;
    if (idx >= NPIX) return;
    int x = idx % IMG_W;
    int y = (idx / IMG_W) % IMG_H;
    int b = idx / (IMG_W * IMG_H);

    // half-pixel bilinear upsample of prev_disp [B, H/2, W/2, 1]
    float sy = (y + 0.5f) * 0.5f - 0.5f;
    float sx = (x + 0.5f) * 0.5f - 0.5f;
    int y0i = (int)floorf(sy); float fy = sy - (float)y0i;
    int x0i = (int)floorf(sx); float fx = sx - (float)x0i;
    int y0c = min(max(y0i, 0), IMG_H / 2 - 1);
    int y1c = min(max(y0i + 1, 0), IMG_H / 2 - 1);
    int x0c = min(max(x0i, 0), IMG_W / 2 - 1);
    int x1c = min(max(x0i + 1, 0), IMG_W / 2 - 1);
    const float* pb = prev + (size_t)b * (IMG_H / 2) * (IMG_W / 2);
    float d00 = pb[y0c * (IMG_W / 2) + x0c];
    float d01 = pb[y0c * (IMG_W / 2) + x1c];
    float d10 = pb[y1c * (IMG_W / 2) + x0c];
    float d11 = pb[y1c * (IMG_W / 2) + x1c];
    float disp = d00 * (1.f - fy) * (1.f - fx) + d01 * (1.f - fy) * fx
               + d10 * fy * (1.f - fx) + d11 * fy * fx;

    // horizontal bilinear warp of `right` at x + disp
    float cx  = (float)x + disp;
    float xf0 = floorf(cx), xf1 = xf0 + 1.f;
    float x0s = fminf(fmaxf(xf0, 0.f), (float)(IMG_W - 1));
    float x1s = fminf(fmaxf(xf1, 0.f), (float)(IMG_W - 1));
    float w0 = (xf1 - cx) * ((xf0 == x0s) ? 1.f : 0.f);
    float w1 = (cx - xf0) * ((xf1 == x1s) ? 1.f : 0.f);
    int i0 = (int)x0s, i1 = (int)x1s;

    const float* rrow = right + ((size_t)(b * IMG_H + y) * IMG_W) * 64;
    const float* lpix = left  + (size_t)idx * 64;
    _Float16* a0 = act0   + (size_t)idx * 96;
    _Float16* wp = warped + (size_t)idx * 64;

    for (int c = 0; c < 64; c += 4) {
        v4f r0 = *(const v4f*)(rrow + (size_t)i0 * 64 + c);
        v4f r1 = *(const v4f*)(rrow + (size_t)i1 * 64 + c);
        v4f lv = *(const v4f*)(lpix + c);
        #pragma unroll
        for (int j = 0; j < 4; ++j) {
            wp[c + j] = (_Float16)(w0 * r0[j] + w1 * r1[j]);
            a0[c + j] = (_Float16)lv[j];
        }
    }
    for (int c = 64; c < 96; ++c) a0[c] = (_Float16)0.f;
}

// ---------------------------------------------------------------------------
// Kernel 2: correlation cost volume, channels 64..68 of act0.
//   cost[i](x) = mean_c left[x][c] * warped[x+i-2][c]   (zero outside row)
// ---------------------------------------------------------------------------
__global__ __launch_bounds__(256)
void prep_cost(const float* __restrict__ left,
               const _Float16* __restrict__ warped,
               _Float16* __restrict__ act0)
{
    int idx = blockIdx.x * 256 + threadIdx.x;
    if (idx >= NPIX) return;
    int x = idx % IMG_W;
    const float* lpix = left + (size_t)idx * 64;

    #pragma unroll
    for (int i = 0; i < 5; ++i) {
        int xx = x + i - 2;
        float c = 0.f;
        if (xx >= 0 && xx < IMG_W) {
            const _Float16* wpix = warped + ((size_t)idx + (i - 2)) * 64;
            for (int ch = 0; ch < 64; ch += 4) {
                v4f lv = *(const v4f*)(lpix + ch);
                c += lv[0] * (float)wpix[ch + 0];
                c += lv[1] * (float)wpix[ch + 1];
                c += lv[2] * (float)wpix[ch + 2];
                c += lv[3] * (float)wpix[ch + 3];
            }
        }
        act0[(size_t)idx * 96 + 64 + i] = (_Float16)(c * (1.f / 64.f));
    }
}

// ---------------------------------------------------------------------------
// Weight conversion: fp32 HWIO [3][3][CinReal][Cout] -> f16 [tap][co][ci_pad]
// (transposed + zero-padded so B-tile loads are contiguous 16B chunks)
// ---------------------------------------------------------------------------
__global__ __launch_bounds__(256)
void cvt_weights(const float* __restrict__ w, _Float16* __restrict__ out,
                 int CinReal, int CinPad, int Cout)
{
    int idx = blockIdx.x * 256 + threadIdx.x;
    int total = 9 * Cout * CinPad;
    if (idx >= total) return;
    int ci  = idx % CinPad;
    int rem = idx / CinPad;
    int co  = rem % Cout;
    int tap = rem / Cout;
    float v = (ci < CinReal) ? w[((size_t)tap * CinReal + ci) * Cout + co] : 0.f;
    out[idx] = (_Float16)v;
}

// ---------------------------------------------------------------------------
// Implicit-GEMM 3x3 SAME conv, f16 in / f32 accum / f16 out, leaky-relu(0.2).
// Block: 256 threads (8 wave32). Tile: 128 pixels (one row segment) x NSUB*16
// output channels. Each wave: 16 pixels x NSUB 16x16 WMMA accumulators.
// K pipeline: 9 taps x KBLK chunks of v_wmma_f32_16x16x32_f16, double-buffered
// LDS (one barrier per K-step; next tile's global loads overlap WMMA work).
// B fragments are software-pipelined (load s+1 before WMMA s) so the compiler
// can use partial s_wait_dscnt and hide LDS latency under WMMA issue.
// LDS rows padded to 40 f16 (80B) to spread banks.
// ---------------------------------------------------------------------------
template<int NSUB, int KBLK>
__global__ __launch_bounds__(256)
void conv3x3_wmma(const _Float16* __restrict__ actIn,
                  const _Float16* __restrict__ wgt,   // [tap][co][CIN] f16
                  const float* __restrict__ bias,
                  _Float16* __restrict__ actOut,
                  int Cout)
{
    constexpr int CIN       = KBLK * 32;
    constexpr int COUT_TILE = NSUB * 16;
    constexpr int LROW      = 40;                  // padded LDS row (elements)
    constexpr int TOT       = 9 * KBLK;
    __shared__ _Float16 lA[2][128 * LROW];
    __shared__ _Float16 lB[2][COUT_TILE * LROW];

    const int tid  = threadIdx.x;
    const int lane = tid & 31;
    const int wave = tid >> 5;
    const int m    = lane & 15;                    // N (or M) index 0..15
    const int half = lane >> 4;                    // lane half select

    const int x0     = blockIdx.x * 128;
    const int y      = blockIdx.y % IMG_H;
    const int b      = blockIdx.y / IMG_H;
    const int coBase = blockIdx.z * COUT_TILE;

    v8f acc[NSUB];
    #pragma unroll
    for (int s = 0; s < NSUB; ++s) { v8f z = {}; acc[s] = z; }

    // global->LDS loader roles
    const int a_px = tid >> 1;                     // 0..127
    const int a_ch = (tid & 1) << 4;               // 0 or 16 (channels)
    const int b_co = tid >> 2;                     // 0..63
    const int b_kq = (tid & 3) << 3;               // 0,8,16,24 (channels)
    const bool okB = (b_co < COUT_TILE);

    auto loadTile = [&](int tap, int kb, v8h& a0, v8h& a1, v8h& bv) {
        const int dy = tap / 3 - 1;
        const int dx = tap % 3 - 1;
        const int yy = y + dy;
        const int xx = x0 + a_px + dx;
        v8h z = {};
        a0 = z; a1 = z; bv = z;
        if (yy >= 0 && yy < IMG_H && xx >= 0 && xx < IMG_W) {
            const _Float16* aSrc = actIn
                + ((size_t)(b * IMG_H + yy) * IMG_W + xx) * CIN + kb * 32 + a_ch;
            a0 = *(const v8h*)(aSrc);
            a1 = *(const v8h*)(aSrc + 8);
        }
        if (okB) {
            bv = *(const v8h*)(wgt
                + ((size_t)(tap * Cout + coBase + b_co)) * CIN + kb * 32 + b_kq);
        }
    };
    auto storeTile = [&](int buf, v8h a0, v8h a1, v8h bv) {
        *(v8h*)&lA[buf][a_px * LROW + a_ch]     = a0;
        *(v8h*)&lA[buf][a_px * LROW + a_ch + 8] = a1;
        if (okB) *(v8h*)&lB[buf][b_co * LROW + b_kq] = bv;
    };

    union V16 { v16h v; v8h h[2]; };

    const int aoff     = (wave * 16 + m) * LROW + half * 16;
    const int boffBase = m * LROW + half * 16;

    // fragment gather + WMMA burst for LDS buffer `cur`; B frags ping-ponged
    auto compute = [&](int cur) {
        V16 af;
        af.h[0] = *(const v8h*)&lA[cur][aoff];
        af.h[1] = *(const v8h*)&lA[cur][aoff + 8];
        V16 bcur, bnxt;
        bcur.h[0] = *(const v8h*)&lB[cur][boffBase];
        bcur.h[1] = *(const v8h*)&lB[cur][boffBase + 8];
        #pragma unroll
        for (int s = 0; s < NSUB; ++s) {
            if (s + 1 < NSUB) {
                bnxt.h[0] = *(const v8h*)&lB[cur][(s + 1) * 16 * LROW + boffBase];
                bnxt.h[1] = *(const v8h*)&lB[cur][(s + 1) * 16 * LROW + boffBase + 8];
            }
            acc[s] = __builtin_amdgcn_wmma_f32_16x16x32_f16(
                false, af.v, false, bcur.v, (short)0, acc[s], false, false);
            bcur = bnxt;
        }
    };

    // prologue: tile 0 into LDS buffer 0
    {
        v8h a0, a1, bv;
        loadTile(0, 0, a0, a1, bv);
        storeTile(0, a0, a1, bv);
    }
    int tapN = 0, kbN = 1;                         // KBLK >= 2 always

    // steady state: branch-free load/compute/store pipeline
    for (int it = 0; it < TOT - 1; ++it) {
        const int cur = it & 1;
        __syncthreads();                           // LDS[cur] writes visible

        v8h na0, na1, nbv;                         // next tile's global loads
        loadTile(tapN, kbN, na0, na1, nbv);        // overlap with WMMA below
        compute(cur);
        storeTile(cur ^ 1, na0, na1, nbv);
        if (++kbN == KBLK) { kbN = 0; ++tapN; }
    }
    __syncthreads();
    compute((TOT - 1) & 1);

    // epilogue: bias + leaky_relu(0.2), f32 -> f16 store
    const size_t rowBase = (size_t)(b * IMG_H + y) * IMG_W;
    #pragma unroll
    for (int s = 0; s < NSUB; ++s) {
        const int co = coBase + s * 16 + m;
        const float bb = bias[co];
        #pragma unroll
        for (int v = 0; v < 8; ++v) {
            const int xg = x0 + wave * 16 + half * 8 + v;
            float val = acc[s][v] + bb;
            val = (val > 0.f) ? val : 0.2f * val;
            actOut[(rowBase + xg) * (size_t)Cout + co] = (_Float16)val;
        }
    }
}

// ---------------------------------------------------------------------------
// Last layer: 3x3 conv, 32 -> 1 channels, linear. Negligible FLOPs; scalar.
// ---------------------------------------------------------------------------
__global__ __launch_bounds__(256)
void conv_last(const _Float16* __restrict__ act5,   // stride 32
               const float* __restrict__ w6,        // [3][3][32][1] fp32
               const float* __restrict__ b6,
               float* __restrict__ out)
{
    int idx = blockIdx.x * 256 + threadIdx.x;
    if (idx >= NPIX) return;
    int x = idx % IMG_W;
    int y = (idx / IMG_W) % IMG_H;
    int b = idx / (IMG_W * IMG_H);

    float acc = b6[0];
    #pragma unroll
    for (int ky = 0; ky < 3; ++ky) {
        int yy = y + ky - 1;
        if (yy < 0 || yy >= IMG_H) continue;
        #pragma unroll
        for (int kx = 0; kx < 3; ++kx) {
            int xx = x + kx - 1;
            if (xx < 0 || xx >= IMG_W) continue;
            const _Float16* ap = act5 + ((size_t)(b * IMG_H + yy) * IMG_W + xx) * 32;
            const float* wp = w6 + (ky * 3 + kx) * 32;
            for (int c = 0; c < 32; ++c) acc += (float)ap[c] * wp[c];
        }
    }
    out[idx] = acc;
}

// ---------------------------------------------------------------------------
// Launch
// ---------------------------------------------------------------------------
extern "C" void kernel_launch(void* const* d_in, const int* in_sizes, int n_in,
                              void* d_out, int out_size, void* d_ws, size_t ws_size,
                              hipStream_t stream)
{
    const float* left  = (const float*)d_in[0];
    const float* right = (const float*)d_in[1];
    const float* prev  = (const float*)d_in[2];
    // d_in[3] = search_range (always 2)
    const float* w1 = (const float*)d_in[4];  const float* b1 = (const float*)d_in[5];
    const float* w2 = (const float*)d_in[6];  const float* b2 = (const float*)d_in[7];
    const float* w3 = (const float*)d_in[8];  const float* b3 = (const float*)d_in[9];
    const float* w4 = (const float*)d_in[10]; const float* b4 = (const float*)d_in[11];
    const float* w5 = (const float*)d_in[12]; const float* b5 = (const float*)d_in[13];
    const float* w6 = (const float*)d_in[14]; const float* b6 = (const float*)d_in[15];
    float* out = (float*)d_out;

    const size_t P = (size_t)NPIX;
    char* ws = (char*)d_ws;
    _Float16* buf0 = (_Float16*)ws;                       // P * 128 f16
    _Float16* buf1 = (_Float16*)(ws + P * 128 * 2);       // P * 128 f16
    _Float16* W1f  = (_Float16*)(ws + 2 * P * 128 * 2);
    _Float16* W2f  = W1f + (size_t)9 * 128 * 96;
    _Float16* W3f  = W2f + (size_t)9 * 128 * 128;
    _Float16* W4f  = W3f + (size_t)9 * 96 * 128;
    _Float16* W5f  = W4f + (size_t)9 * 64 * 96;

    const int pixBlocks = (NPIX + 255) / 256;

    // weight conversions (independent of activations)
    cvt_weights<<<(9 * 128 * 96  + 255) / 256, 256, 0, stream>>>(w1, W1f,  69,  96, 128);
    cvt_weights<<<(9 * 128 * 128 + 255) / 256, 256, 0, stream>>>(w2, W2f, 128, 128, 128);
    cvt_weights<<<(9 * 96  * 128 + 255) / 256, 256, 0, stream>>>(w3, W3f, 128, 128,  96);
    cvt_weights<<<(9 * 64  * 96  + 255) / 256, 256, 0, stream>>>(w4, W4f,  96,  96,  64);
    cvt_weights<<<(9 * 32  * 64  + 255) / 256, 256, 0, stream>>>(w5, W5f,  64,  64,  32);

    // warp + cost volume -> act0 (buf0, stride 96); warped lives in buf1
    prep_warp<<<pixBlocks, 256, 0, stream>>>(left, right, prev, buf0, buf1);
    prep_cost<<<pixBlocks, 256, 0, stream>>>(left, buf1, buf0);

    const dim3 blk(256);
    const dim3 g2(IMG_W / 128, IMG_B * IMG_H, 2);
    const dim3 g1(IMG_W / 128, IMG_B * IMG_H, 1);

    // L1:  96 -> 128   (2 tiles of 64 couts)
    conv3x3_wmma<4, 3><<<g2, blk, 0, stream>>>(buf0, W1f, b1, buf1, 128);
    // L2: 128 -> 128
    conv3x3_wmma<4, 4><<<g2, blk, 0, stream>>>(buf1, W2f, b2, buf0, 128);
    // L3: 128 ->  96   (2 tiles of 48 couts)
    conv3x3_wmma<3, 4><<<g2, blk, 0, stream>>>(buf0, W3f, b3, buf1,  96);
    // L4:  96 ->  64
    conv3x3_wmma<4, 3><<<g1, blk, 0, stream>>>(buf1, W4f, b4, buf0,  64);
    // L5:  64 ->  32
    conv3x3_wmma<2, 2><<<g1, blk, 0, stream>>>(buf0, W5f, b5, buf1,  32);
    // L6:  32 ->   1 (linear)
    conv_last<<<pixBlocks, 256, 0, stream>>>(buf1, w6, b6, out);
}
